// GraphSPADEGeneratorUnit_88510686036719
// MI455X (gfx1250) — matrix operand order repaired and checked by reading, not compiled, not run on hardware
//
#include <hip/hip_runtime.h>
#include <hip/hip_bf16.h>

// ---------------------------------------------------------------------------
// GraphSPADE generator forward for MI455X (gfx1250, wave32, WMMA).
// Dense GEMMs -> v_wmma_f32_16x16x32_f16; fully compile-time tile shapes
// (K and M templated), unconditional float4 operand loads, unconditional
// immediate-offset stores on the full-tile path.
// Graph aggregation -> coalesced gathers + f32 atomics (L2-resident arrays).
// ---------------------------------------------------------------------------

#define LRELU_NEG 0.2f
#define BN_EPS    1e-5f

typedef __attribute__((ext_vector_type(16))) _Float16 v16h;
typedef __attribute__((ext_vector_type(8)))  float    v8f;

__device__ __forceinline__ float lrelu(float v) { return v > 0.0f ? v : LRELU_NEG * v; }

// float atomic max via ordered-int trick (valid for all non-NaN floats)
__device__ __forceinline__ void atomicMaxF(float* addr, float val) {
  if (val >= 0.0f) atomicMax((int*)addr, __float_as_int(val));
  else             atomicMin((unsigned int*)addr, __float_as_uint(val));
}

// compile-time-indexed f32x4 -> f16 insert (avoids dynamic vector indexing)
template <int B>
__device__ __forceinline__ void cvt4(v16h& v, float4 f) {
  v[B + 0] = (_Float16)f.x;
  v[B + 1] = (_Float16)f.y;
  v[B + 2] = (_Float16)f.z;
  v[B + 3] = (_Float16)f.w;
}

// ---------------------------------------------------------------------------
// WMMA GEMM: C[Nr x M] = A[Nr x K] @ W[K x M], W given transposed as
// Wt[M x K]. K, M compile-time; K multiple of 32; one 16x16 tile per wave32.
// A-operand layout: lane holds row (lane&15); element e -> K index
//   (e<8 ? 8h+e : 16+8h+(e-8)), h = lane>>4  => two runs of 8 floats.
// B-operand layout: lane holds col (lane&15); element e -> K index
//   (e<8 ? 8h+e : 16+8h+(e-8)) => two runs of 8 floats from Wt.
// ---------------------------------------------------------------------------
template <int K, int M>
__global__ void gemm_wmma_f16(const float* __restrict__ A,
                              const float* __restrict__ Wt,
                              float* __restrict__ C,
                              int Nr) {
  constexpr int Mt = M >> 4;
  const int wid  = (blockIdx.x * blockDim.x + threadIdx.x) >> 5;
  const int lane = threadIdx.x & 31;
  const int Nt   = (Nr + 15) >> 4;
  if (wid >= Nt * Mt) return;                 // wave-uniform: EXEC all-ones below

  const int row0 = (wid / Mt) << 4;
  const int col0 = (wid % Mt) << 4;
  const int h    = lane >> 4;                 // lane half (0/1)
  const int mn   = lane & 15;                 // row (A) / col (B,C) within tile

  int arow = row0 + mn;
  if (arow >= Nr) arow = Nr - 1;              // clamp (no predicated loads)
  const float4* A4 = (const float4*)(A + (size_t)arow * K);
  const float4* B4 = (const float4*)(Wt + (size_t)(col0 + mn) * K);

  v8f acc = {};
#pragma unroll
  for (int k0 = 0; k0 < K; k0 += 32) {
    const int q = k0 >> 2;
    float4 a0 = A4[q + 2 * h];                // k = k0+8h   .. +3
    float4 a1 = A4[q + 2 * h + 1];            // k = k0+8h+4 .. +7
    float4 a2 = A4[q + 4 + 2 * h];            // k = k0+16+8h ..
    float4 a3 = A4[q + 4 + 2 * h + 1];
    float4 b0 = B4[q + 2 * h];                // k = k0+8h .. +3
    float4 b1 = B4[q + 2 * h + 1];
    float4 b2 = B4[q + 4 + 2 * h];
    float4 b3 = B4[q + 4 + 2 * h + 1];

    v16h a, b;
    cvt4<0>(a, a0);  cvt4<4>(a, a1);  cvt4<8>(a, a2);  cvt4<12>(a, a3);
    cvt4<0>(b, b0);  cvt4<4>(b, b1);  cvt4<8>(b, b2);  cvt4<12>(b, b3);

    acc = __builtin_amdgcn_wmma_f32_16x16x32_f16(false, a, false, b,
                                                 (short)0, acc, false, false);
  }

  // C f32 layout: VGPR r -> row row0 + 8h + r, col = col0 + mn.
  float* Crow = C + (size_t)(row0 + 8 * h) * M + col0 + mn;
  if (row0 + 16 <= Nr) {                      // full tile: unconditional stores
#pragma unroll
    for (int r = 0; r < 8; ++r) Crow[r * M] = acc[r];
  } else {                                    // tail tile (not hit when Nr%16==0)
#pragma unroll
    for (int r = 0; r < 8; ++r)
      if (row0 + 8 * h + r < Nr) Crow[r * M] = acc[r];
  }
}

// Wt[m*Kpad + k] = (k < Korig) ? W[k*M + m] : 0
__global__ void k_transpose_pad(const float* __restrict__ W, float* __restrict__ Wt,
                                int Korig, int Kpad, int M) {
  int idx = blockIdx.x * blockDim.x + threadIdx.x;
  if (idx >= M * Kpad) return;
  int m = idx / Kpad, k = idx % Kpad;
  Wt[idx] = (k < Korig) ? W[(size_t)k * M + m] : 0.0f;
}

// maskp[N x 32]: cols 0..15 = mask, 16..31 = 0
__global__ void k_pad_mask(const float* __restrict__ mask, float* __restrict__ mp, int N) {
  long idx = (long)blockIdx.x * blockDim.x + threadIdx.x;
  if (idx >= (long)N * 32) return;
  int i = (int)(idx >> 5), c = (int)(idx & 31);
  mp[idx] = (c < 16) ? mask[(size_t)i * 16 + c] : 0.0f;
}

// ---------------------------------------------------------------------------
// Graph / elementwise kernels
// ---------------------------------------------------------------------------
__global__ void k_fill(float* p, float v, long n) {
  long i = (long)blockIdx.x * blockDim.x + threadIdx.x;
  if (i < n) p[i] = v;
}

__global__ void k_deg(const int* __restrict__ dst, float* deg, int E) {
  for (int e = blockIdx.x * blockDim.x + threadIdx.x; e < E; e += gridDim.x * blockDim.x)
    atomicAdd(&deg[dst[e]], 1.0f);
}

__global__ void k_rsqrt_inplace(float* d, int n) {
  int i = blockIdx.x * blockDim.x + threadIdx.x;
  if (i < n) d[i] = rsqrtf(d[i]);             // deg >= 1 always (self-loop)
}

__global__ void k_norm(const int* __restrict__ src, const int* __restrict__ dst,
                       const float* __restrict__ dinv, float* nrm, int E) {
  for (int e = blockIdx.x * blockDim.x + threadIdx.x; e < E; e += gridDim.x * blockDim.x)
    nrm[e] = dinv[src[e]] * dinv[dst[e]];
}

// out[i,f] = bias[f] + dinv[i]^2 * h[i,f]   (self-loop term, no atomics)
__global__ void k_gcn_init(const float* __restrict__ h, const float* __restrict__ bias,
                           const float* __restrict__ dinv, float* out, int N, int F) {
  int f = threadIdx.x;
  for (int i = blockIdx.x; i < N; i += gridDim.x) {
    float w = dinv[i]; w *= w;
    out[(size_t)i * F + f] = bias[f] + w * h[(size_t)i * F + f];
  }
}

// out[dst,f] += norm[e] * h[src,f]
__global__ void k_gcn_scatter(const float* __restrict__ h, const int* __restrict__ src,
                              const int* __restrict__ dst, const float* __restrict__ nrm,
                              float* out, int E, int F) {
  int f = threadIdx.x;
  for (int e = blockIdx.x; e < E; e += gridDim.x) {
    int s = src[e], d = dst[e];
    atomicAdd(&out[(size_t)d * F + f], nrm[e] * h[(size_t)s * F + f]);
  }
}

__global__ void k_relu_inplace(float* p, long n) {
  long i = (long)blockIdx.x * blockDim.x + threadIdx.x;
  if (i < n) p[i] = fmaxf(p[i], 0.0f);
}

// column sums / sums-of-squares for 64-wide BN (atomic partials)
__global__ void k_bn_stats(const float* __restrict__ x, float* sum, float* sumsq, int N) {
  int c = threadIdx.x & 63;
  int r0 = threadIdx.x >> 6;                  // 256 threads -> 4 row-lanes
  float s = 0.0f, q = 0.0f;
  for (int i = blockIdx.x * 4 + r0; i < N; i += gridDim.x * 4) {
    float v = x[(size_t)i * 64 + c];
    s += v; q += v * v;
  }
  atomicAdd(&sum[c], s);
  atomicAdd(&sumsq[c], q);
}

// stats: [0:64)=sum [64:128)=sumsq [128:192)=mu [192:256)=rstd
__global__ void k_bn_final(float* stats, int N) {
  int c = threadIdx.x;
  float mu  = stats[c] / (float)N;
  float var = stats[64 + c] / (float)N - mu * mu;
  stats[128 + c] = mu;
  stats[192 + c] = rsqrtf(var + BN_EPS);
}

// out = lrelu( bn(x) * (1+gamma) + beta )
__global__ void k_spade_combine(const float* __restrict__ x, const float* __restrict__ gma,
                                const float* __restrict__ bta, const float* __restrict__ stats,
                                float* out, int N) {
  long idx = (long)blockIdx.x * blockDim.x + threadIdx.x;
  if (idx >= (long)N * 64) return;
  int c = (int)(idx & 63);
  float xn = (x[idx] - stats[128 + c]) * stats[192 + c];
  float v = xn * (1.0f + gma[idx]) + bta[idx];
  out[idx] = lrelu(v);
}

// es[i,hd] = <h[i,hd,:], a_s[hd,:]> ; ed likewise
__global__ void k_gat_scores(const float* __restrict__ h, const float* __restrict__ as,
                             const float* __restrict__ ad, float* es, float* ed, int N) {
  long t = (long)blockIdx.x * blockDim.x + threadIdx.x;
  if (t >= (long)N * 4) return;
  int i = (int)(t >> 2), hd = (int)(t & 3);
  const float* hp = h + (size_t)i * 64 + hd * 16;
  float s = 0.0f, d = 0.0f;
#pragma unroll
  for (int c = 0; c < 16; ++c) {
    float v = hp[c];
    s += v * as[hd * 16 + c];
    d += v * ad[hd * 16 + c];
  }
  es[t] = s; ed[t] = d;
}

// m init with self-loop score
__global__ void k_gat_init_m(const float* __restrict__ es, const float* __restrict__ ed,
                             float* m, long n4) {
  long t = (long)blockIdx.x * blockDim.x + threadIdx.x;
  if (t < n4) m[t] = lrelu(es[t] + ed[t]);
}

// segment max over edges
__global__ void k_gat_edge_max(const int* __restrict__ src, const int* __restrict__ dst,
                               const float* __restrict__ es, const float* __restrict__ ed,
                               float* m, int E) {
  for (int e = blockIdx.x * blockDim.x + threadIdx.x; e < E; e += gridDim.x * blockDim.x) {
    int s4 = src[e] * 4, d4 = dst[e] * 4;
#pragma unroll
    for (int hd = 0; hd < 4; ++hd)
      atomicMaxF(&m[d4 + hd], lrelu(es[s4 + hd] + ed[d4 + hd]));
  }
}

// self-loop contributions: s[i,hd] = exp(e_self - m); ou[i,f] = p_self * h[i,f]
__global__ void k_gat_init_s(const float* __restrict__ es, const float* __restrict__ ed,
                             const float* __restrict__ m, const float* __restrict__ h,
                             float* s, float* ou, int N) {
  int f = threadIdx.x;                        // 64 threads
  int hd = f >> 4;
  for (int i = blockIdx.x; i < N; i += gridDim.x) {
    float ev = lrelu(es[i * 4 + hd] + ed[i * 4 + hd]);
    float p = __expf(ev - m[i * 4 + hd]);
    if ((f & 15) == 0) s[i * 4 + hd] = p;
    ou[(size_t)i * 64 + f] = p * h[(size_t)i * 64 + f];
  }
}

// edges: p = exp(lrelu(es[src]+ed[dst]) - m[dst]); s[dst]+=p; ou[dst,f]+=p*h[src,f]
__global__ void k_gat_edge_scatter(const int* __restrict__ src, const int* __restrict__ dst,
                                   const float* __restrict__ es, const float* __restrict__ ed,
                                   const float* __restrict__ m, const float* __restrict__ h,
                                   float* s, float* ou, int E) {
  int f = threadIdx.x;                        // 64 threads
  int hd = f >> 4;
  for (int e = blockIdx.x; e < E; e += gridDim.x) {
    int ss = src[e], dd = dst[e];
    float ev = lrelu(es[ss * 4 + hd] + ed[dd * 4 + hd]);
    float p = __expf(ev - m[dd * 4 + hd]);
    if ((f & 15) == 0) atomicAdd(&s[dd * 4 + hd], p);
    atomicAdd(&ou[(size_t)dd * 64 + f], p * h[(size_t)ss * 64 + f]);
  }
}

// out = ou / s + bias (+ addend)
__global__ void k_gat_final(const float* __restrict__ ou, const float* __restrict__ s,
                            const float* __restrict__ b, const float* __restrict__ addend,
                            float* out, int N) {
  long idx = (long)blockIdx.x * blockDim.x + threadIdx.x;
  if (idx >= (long)N * 64) return;
  int i = (int)(idx >> 6), f = (int)(idx & 63), hd = f >> 4;
  float v = ou[idx] / s[i * 4 + hd] + b[f];
  if (addend) v += addend[idx];
  out[idx] = v;
}

// ---------------------------------------------------------------------------
// Host-side composition
// ---------------------------------------------------------------------------
// A row stride must equal Kpad. Transposes/pads W into wt, then dispatches.
static void run_gemm(const float* A, const float* W, float* C,
                     int N, int Korig, int Kpad, int M, float* wt, hipStream_t st) {
  int tot = M * Kpad;
  k_transpose_pad<<<(tot + 255) / 256, 256, 0, st>>>(W, wt, Korig, Kpad, M);
  int tiles = ((N + 15) / 16) * (M / 16);
  int blocks = (tiles + 7) / 8;               // 8 waves (256 thr) per block
  if (Kpad == 32 && M == 128)
    gemm_wmma_f16<32, 128><<<blocks, 256, 0, st>>>(A, wt, C, N);
  else if (Kpad == 128 && M == 64)
    gemm_wmma_f16<128, 64><<<blocks, 256, 0, st>>>(A, wt, C, N);
  else // Kpad == 64, M == 64
    gemm_wmma_f16<64, 64><<<blocks, 256, 0, st>>>(A, wt, C, N);
}

static void run_gcn(const float* Xin, const float* W, const float* bias,
                    int Korig, int Kpad, int F,
                    float* tmp, float* out, float* wt,
                    const float* dinv, const float* nrm,
                    const int* src, const int* dst, int N, int E, hipStream_t st) {
  run_gemm(Xin, W, tmp, N, Korig, Kpad, F, wt, st);
  k_gcn_init<<<2048, F, 0, st>>>(tmp, bias, dinv, out, N, F);
  k_gcn_scatter<<<8192, F, 0, st>>>(tmp, src, dst, nrm, out, E, F);
}

extern "C" void kernel_launch(void* const* d_in, const int* in_sizes, int n_in,
                              void* d_out, int out_size, void* d_ws, size_t ws_size,
                              hipStream_t stream) {
  const float* x    = (const float*)d_in[0];
  const float* mask = (const float*)d_in[1];
  const int*   ei   = (const int*)d_in[2];
  const int N = in_sizes[0] / 64;
  const int E = in_sizes[2] / 2;
  const int* src = ei;
  const int* dst = ei + E;

  // SPADE params: (w0,b0,wg,bg,wb,bb) x3 at indices 3..20; GAT at 21..32
  const float* sp[3][6];
  for (int p = 0; p < 3; ++p)
    for (int j = 0; j < 6; ++j) sp[p][j] = (const float*)d_in[3 + p * 6 + j];
  const float* gp[3][4];
  for (int g = 0; g < 3; ++g)
    for (int j = 0; j < 4; ++j) gp[g][j] = (const float*)d_in[21 + g * 4 + j];

  // ---- workspace carve (floats; all block sizes keep 16B alignment) ----
  float* Wf = (float*)d_ws;
  size_t o = 0;
  auto alloc = [&](size_t n) { float* p = Wf + o; o += n; return p; };
  float* dinv = alloc(N);
  float* nrm  = alloc(E);
  float* mkp  = alloc((size_t)N * 32);    // zero-padded mask [N,32]
  float* t128 = alloc((size_t)N * 128);   // GEMM output before aggregation
  float* mf   = alloc((size_t)N * 128);
  float* gma  = alloc((size_t)N * 64);
  float* bta  = alloc((size_t)N * 64);
  float* X1   = alloc((size_t)N * 64);
  float* X2   = alloc((size_t)N * 64);
  float* gh   = alloc((size_t)N * 64);
  float* ou   = alloc((size_t)N * 64);
  float* es   = alloc((size_t)N * 4);
  float* ed   = alloc((size_t)N * 4);
  float* mm   = alloc((size_t)N * 4);
  float* sm   = alloc((size_t)N * 4);
  float* wt   = alloc(16384);             // transposed/padded weight staging
  float* stats = alloc(256);
  (void)ws_size; (void)n_in; (void)out_size;

  // ---- degrees / normalization (self-loops folded in: deg init = 1) ----
  k_fill<<<(N + 255) / 256, 256, 0, stream>>>(dinv, 1.0f, N);
  k_deg<<<2048, 256, 0, stream>>>(dst, dinv, E);
  k_rsqrt_inplace<<<(N + 255) / 256, 256, 0, stream>>>(dinv, N);
  k_norm<<<2048, 256, 0, stream>>>(src, dst, dinv, nrm, E);
  k_pad_mask<<<(int)(((long)N * 32 + 255) / 256), 256, 0, stream>>>(mask, mkp, N);

  long n64 = (long)N * 64;
  int  g64 = (int)((n64 + 255) / 256);
  int  g4  = (int)(((long)N * 4 + 255) / 256);

  // ---- SPADE block ----
  auto spade = [&](const float* xin, const float* const* P, float* out) {
    // mf = relu(gcn(mask, w0, b0))  [N,128]   (mask padded to K=32)
    run_gcn(mkp, P[0], P[1], 16, 32, 128, t128, mf, wt, dinv, nrm, src, dst, N, E, stream);
    k_relu_inplace<<<(int)(((long)N * 128 + 255) / 256), 256, 0, stream>>>(mf, (long)N * 128);
    // gamma / beta  [N,64]
    run_gcn(mf, P[2], P[3], 128, 128, 64, t128, gma, wt, dinv, nrm, src, dst, N, E, stream);
    run_gcn(mf, P[4], P[5], 128, 128, 64, t128, bta, wt, dinv, nrm, src, dst, N, E, stream);
    // batchnorm stats of xin
    k_fill<<<1, 128, 0, stream>>>(stats, 0.0f, 128);
    k_bn_stats<<<1024, 256, 0, stream>>>(xin, stats, stats + 64, N);
    k_bn_final<<<1, 64, 0, stream>>>(stats, N);
    // combine + leaky_relu
    k_spade_combine<<<g64, 256, 0, stream>>>(xin, gma, bta, stats, out, N);
  };

  // ---- GAT block ----
  auto gat = [&](const float* xin, const float* const* P, const float* addend, float* out) {
    run_gemm(xin, P[0], gh, N, 64, 64, 64, wt, stream);              // h = x @ W
    k_gat_scores<<<g4, 256, 0, stream>>>(gh, P[2], P[3], es, ed, N);
    k_gat_init_m<<<g4, 256, 0, stream>>>(es, ed, mm, (long)N * 4);
    k_gat_edge_max<<<2048, 256, 0, stream>>>(src, dst, es, ed, mm, E);
    k_gat_init_s<<<4096, 64, 0, stream>>>(es, ed, mm, gh, sm, ou, N);
    k_gat_edge_scatter<<<8192, 64, 0, stream>>>(src, dst, es, ed, mm, gh, sm, ou, E);
    k_gat_final<<<g64, 256, 0, stream>>>(ou, sm, P[1], addend, out, N);
  };

  // ---- forward ----
  spade(x,  sp[0], X1);              // xm = lrelu(spade_s1(x))
  gat(X1, gp[0], nullptr, X2);       // xm = gat_g1(xm)
  spade(X2, sp[1], X1);              // xm = lrelu(spade_s2(xm))
  gat(X1, gp[1], nullptr, X2);       // xm = gat_g2(xm)
  spade(x,  sp[2], X1);              // xs = lrelu(spade_sk(x))
  gat(X1, gp[2], X2, (float*)d_out); // out = gat_gk(xs) + xm
}
